// GateLoopLayer_35768487641311
// MI455X (gfx1250) — matrix operand order repaired and checked by reading, not compiled, not run on hardware
//
#include <hip/hip_runtime.h>
#include <math.h>

#define SEQ 1024
#define DIM 256
#define GROUPS 64
#define EPS 1e-5f

typedef __attribute__((ext_vector_type(16))) __bf16 v16bf;
typedef __attribute__((ext_vector_type(8)))  __bf16 v8bf;
typedef __attribute__((ext_vector_type(8)))  float  v8f;

#if __has_builtin(__builtin_amdgcn_global_load_async_to_lds_b32) && \
    __has_builtin(__builtin_amdgcn_s_wait_asynccnt)
#define USE_ASYNC_LDS 1
typedef __attribute__((address_space(1))) int g_int;
typedef __attribute__((address_space(3))) int l_int;
#else
#define USE_ASYNC_LDS 0
#endif

// ---------------------------------------------------------------------------
// f32 -> bf16 conversion (activations)
// ---------------------------------------------------------------------------
__global__ __launch_bounds__(256)
void f32_to_bf16_kernel(const float* __restrict__ src, __bf16* __restrict__ dst, int n) {
    int i = blockIdx.x * 256 + threadIdx.x;
    if (i < n) dst[i] = (__bf16)src[i];
}

// ---------------------------------------------------------------------------
// f32 -> bf16 with transpose: src is R x C row-major, dst is C x R row-major.
// Weights are transposed once so WMMA B-fragments become contiguous loads.
// ---------------------------------------------------------------------------
__global__ __launch_bounds__(256)
void f32_to_bf16_T_kernel(const float* __restrict__ src, __bf16* __restrict__ dst,
                          int R, int C) {
    int i = blockIdx.x * 256 + threadIdx.x;
    if (i >= R * C) return;
    int r = i / C, c = i - r * C;
    dst[c * R + r] = (__bf16)src[i];
}

// ---------------------------------------------------------------------------
// WMMA GEMM: C(MxN, f32) = A(MxK, bf16 row-major) * BT(NxK, bf16 row-major)^T
// [+ bias]. One 16x16 output tile per wave; K consumed 32 at a time with
// v_wmma_f32_16x16x32_bf16. Fragment layouts per CDNA5 ISA 7.12.2; all
// fragment loads are contiguous 16B runs (global_load_b128).
// ---------------------------------------------------------------------------
__global__ __launch_bounds__(256)
void gemm_bf16_wmma_kernel(const __bf16* __restrict__ A, const __bf16* __restrict__ BT,
                           const float* __restrict__ bias, float* __restrict__ C,
                           int M, int N, int K) {
    const int lane = threadIdx.x & 31;
    const int wave = threadIdx.x >> 5;
    const int tilesN = N >> 4;
    const int totalTiles = (M >> 4) * tilesN;
    const int tile = blockIdx.x * 8 + wave;       // wave-uniform
    if (tile >= totalTiles) return;               // uniform branch: EXEC stays all-1s
    const int m0 = (tile / tilesN) << 4;
    const int n0 = (tile % tilesN) << 4;

    const int rowA   = m0 + (lane & 15);
    const int khalfA = (lane >> 4) * 8;           // A: lanes 16-31 hold K+8 block
    const int colB   = n0 + (lane & 15);
    const int kbaseB = (lane >> 4) * 8;           // B: lanes 16-31 hold K+8 block

    v8f acc = {};
    for (int kc = 0; kc < K; kc += 32) {
        const __bf16* ap = A + rowA * K + kc + khalfA;
        v8bf alo = *(const v8bf*)(ap);            // K = khalf + 0..7
        v8bf ahi = *(const v8bf*)(ap + 16);       // K = khalf + 16..23
        v16bf a = __builtin_shufflevector(alo, ahi,
                   0, 1, 2, 3, 4, 5, 6, 7, 8, 9, 10, 11, 12, 13, 14, 15);

        const __bf16* bp = BT + colB * K + kc + kbaseB;
        v8bf blo = *(const v8bf*)(bp);            // K = kbase + 0..7
        v8bf bhi = *(const v8bf*)(bp + 16);       // K = kbase + 16..23
        v16bf b = __builtin_shufflevector(blo, bhi,
                   0, 1, 2, 3, 4, 5, 6, 7, 8, 9, 10, 11, 12, 13, 14, 15);

        acc = __builtin_amdgcn_wmma_f32_16x16x32_bf16(
            /*neg_a=*/false, a, /*neg_b=*/false, b,
            /*c_mod=*/(short)0, acc, /*reuse_a=*/false, /*reuse_b=*/false);
    }
#pragma unroll
    for (int r = 0; r < 8; ++r) {                 // C/D: vgpr r -> row r (+8 for hi lanes)
        int row = m0 + r + ((lane >> 4) << 3);
        int col = n0 + (lane & 15);
        float v = acc[r];
        if (bias) v += bias[col];
        C[row * N + col] = v;
    }
}

// ---------------------------------------------------------------------------
// a_complex = sigmoid(a_real) * exp(i*a_imag) -> (amr, ami)
// ---------------------------------------------------------------------------
__global__ __launch_bounds__(256)
void gate_decompose_kernel(const float* __restrict__ af,
                           float* __restrict__ amr, float* __restrict__ ami) {
    int i = blockIdx.x * 256 + threadIdx.x;       // 0 .. SEQ*DIM-1
    if (i >= SEQ * DIM) return;
    int l = i >> 8, e = i & (DIM - 1);
    float ar = af[l * (2 * DIM) + e];
    float ai = af[l * (2 * DIM) + DIM + e];
    float s  = 1.0f / (1.0f + __expf(-ar));
    float sn, cs;
    __sincosf(ai, &sn, &cs);
    amr[i] = s * cs;
    ami[i] = s * sn;
}

// ---------------------------------------------------------------------------
// Sequential complex gated scan. Workgroup b owns d in [16b, 16b+16);
// thread t owns column e=t with 16 complex state elements in registers.
// Per step: S[d] = a[d]*S[d] + k[d]*v[e]; c = sum_d q[d]*Re(S[d]).
// Per-d scalars staged in LDS in 32-step chunks; staging uses the gfx1250
// async-to-LDS path (ASYNCcnt) when the builtins are available.
// ---------------------------------------------------------------------------
__global__ __launch_bounds__(256)
void gated_scan_kernel(const float* __restrict__ kf, const float* __restrict__ vf,
                       const float* __restrict__ qf, const float* __restrict__ amr,
                       const float* __restrict__ ami, float* __restrict__ part) {
    __shared__ float sK[32][16], sQ[32][16], sAr[32][16], sAi[32][16];
    const int t  = threadIdx.x;       // e column
    const int d0 = blockIdx.x * 16;   // d block
    float sre[16], sim[16];
#pragma unroll
    for (int j = 0; j < 16; ++j) { sre[j] = 0.0f; sim[j] = 0.0f; }
    float* pout = part + (size_t)blockIdx.x * SEQ * DIM;

    for (int l0 = 0; l0 < SEQ; l0 += 32) {
        __syncthreads();
#if USE_ASYNC_LDS
        for (int i = t; i < 512; i += 256) {
            int s = i >> 4, j = i & 15;
            int g = (l0 + s) * DIM + d0 + j;
            __builtin_amdgcn_global_load_async_to_lds_b32(
                (g_int*)(kf + g),  (l_int*)(&sK[0][0] + i), 0, 0);
            __builtin_amdgcn_global_load_async_to_lds_b32(
                (g_int*)(qf + g),  (l_int*)(&sQ[0][0] + i), 0, 0);
            __builtin_amdgcn_global_load_async_to_lds_b32(
                (g_int*)(amr + g), (l_int*)(&sAr[0][0] + i), 0, 0);
            __builtin_amdgcn_global_load_async_to_lds_b32(
                (g_int*)(ami + g), (l_int*)(&sAi[0][0] + i), 0, 0);
        }
        __builtin_amdgcn_s_wait_asynccnt(0);
#else
        for (int i = t; i < 512; i += 256) {
            int s = i >> 4, j = i & 15;
            int g = (l0 + s) * DIM + d0 + j;
            sK[s][j]  = kf[g];
            sQ[s][j]  = qf[g];
            sAr[s][j] = amr[g];
            sAi[s][j] = ami[g];
        }
#endif
        __syncthreads();
        if (l0 + 32 < SEQ) {          // gfx1250 global_prefetch_b8 for next chunk
            __builtin_prefetch(&kf[(l0 + 32) * DIM + d0], 0, 1);
            __builtin_prefetch(&qf[(l0 + 32) * DIM + d0], 0, 1);
            __builtin_prefetch(&amr[(l0 + 32) * DIM + d0], 0, 1);
            __builtin_prefetch(&ami[(l0 + 32) * DIM + d0], 0, 1);
        }
        for (int s = 0; s < 32; ++s) {
            int l = l0 + s;
            float vr = vf[l * DIM + t];
            float c = 0.0f;
#pragma unroll
            for (int j = 0; j < 16; ++j) {
                float ar = sAr[s][j], ai = sAi[s][j];
                float re = fmaf(ar, sre[j], fmaf(-ai, sim[j], sK[s][j] * vr));
                float im = fmaf(ar, sim[j], ai * sre[j]);
                sre[j] = re; sim[j] = im;
                c = fmaf(sQ[s][j], re, c);
            }
            pout[l * DIM + t] = c;
        }
    }
}

// ---------------------------------------------------------------------------
// Reduce 16 partial blocks -> y, GroupNorm(64 groups of 4), * silu(g),
// emit bf16 z for the final WMMA GEMM. One thread per (l, group).
// ---------------------------------------------------------------------------
__global__ __launch_bounds__(256)
void reduce_gn_gate_kernel(const float* __restrict__ part, const float* __restrict__ gf,
                           const float* __restrict__ gnw, const float* __restrict__ gnb,
                           __bf16* __restrict__ zb) {
    int idx = blockIdx.x * 256 + threadIdx.x;     // 0 .. SEQ*GROUPS-1
    if (idx >= SEQ * GROUPS) return;
    int l = idx >> 6, g = idx & (GROUPS - 1);
    float ych[4];
#pragma unroll
    for (int c = 0; c < 4; ++c) {
        int e = 4 * g + c;
        float s = 0.0f;
#pragma unroll
        for (int b = 0; b < 16; ++b)
            s += part[(size_t)b * SEQ * DIM + l * DIM + e];
        ych[c] = s;
    }
    float mean = 0.25f * (ych[0] + ych[1] + ych[2] + ych[3]);
    float var = 0.0f;
#pragma unroll
    for (int c = 0; c < 4; ++c) { float d = ych[c] - mean; var = fmaf(d, d, var); }
    var *= 0.25f;
    float inv = rsqrtf(var + EPS);
#pragma unroll
    for (int c = 0; c < 4; ++c) {
        int e = 4 * g + c;
        float gn = (ych[c] - mean) * inv * gnw[e] + gnb[e];
        float gv = gf[l * DIM + e];
        float z  = gn * (gv / (1.0f + __expf(-gv)));   // * silu(g)
        zb[l * DIM + e] = (__bf16)z;
    }
}

// ---------------------------------------------------------------------------
// Host-side orchestration
// ---------------------------------------------------------------------------
static inline int cdiv(int a, int b) { return (a + b - 1) / b; }

extern "C" void kernel_launch(void* const* d_in, const int* in_sizes, int n_in,
                              void* d_out, int out_size, void* d_ws, size_t ws_size,
                              hipStream_t stream) {
    const float* x   = (const float*)d_in[0];   // (1,1024,256)
    const float* wq  = (const float*)d_in[1];   // (256,256)
    const float* wk  = (const float*)d_in[2];
    const float* wv  = (const float*)d_in[3];
    const float* wa  = (const float*)d_in[4];   // (256,512)
    const float* wg  = (const float*)d_in[5];
    const float* wo  = (const float*)d_in[6];
    const float* gnw = (const float*)d_in[7];
    const float* gnb = (const float*)d_in[8];
    const float* bias= (const float*)d_in[9];
    float* out = (float*)d_out;

    char* ws = (char*)d_ws;
    // workspace layout (bytes); weight buffers hold transposed bf16 copies
    __bf16* xb  = (__bf16*)(ws + 0);            // 1024*256*2   = 524288
    __bf16* wqb = (__bf16*)(ws + 524288);       // 256*256*2    = 131072
    __bf16* wkb = (__bf16*)(ws + 655360);
    __bf16* wvb = (__bf16*)(ws + 786432);
    __bf16* wgb = (__bf16*)(ws + 917504);
    __bf16* wab = (__bf16*)(ws + 1048576);      // 512*256*2    = 262144
    __bf16* wob = (__bf16*)(ws + 1310720);      // 131072
    float*  qf  = (float*)(ws + 1441792);       // 1024*256*4   = 1048576
    float*  kf  = (float*)(ws + 2490368);
    float*  vf  = (float*)(ws + 3538944);
    float*  gf  = (float*)(ws + 4587520);
    float*  af  = (float*)(ws + 5636096);       // 1024*512*4   = 2097152
    float*  amr = (float*)(ws + 7733248);       // 1048576
    float*  ami = (float*)(ws + 8781824);       // 1048576
    float*  part= (float*)(ws + 9830400);       // 16*1024*256*4 = 16777216
    __bf16* zb  = (__bf16*)(ws + 26607616);     // 524288

    // 1) convert x to bf16; convert+transpose weights to bf16
    f32_to_bf16_kernel<<<cdiv(SEQ * DIM, 256), 256, 0, stream>>>(x, xb, SEQ * DIM);
    f32_to_bf16_T_kernel<<<cdiv(DIM * DIM, 256), 256, 0, stream>>>(wq, wqb, DIM, DIM);
    f32_to_bf16_T_kernel<<<cdiv(DIM * DIM, 256), 256, 0, stream>>>(wk, wkb, DIM, DIM);
    f32_to_bf16_T_kernel<<<cdiv(DIM * DIM, 256), 256, 0, stream>>>(wv, wvb, DIM, DIM);
    f32_to_bf16_T_kernel<<<cdiv(DIM * DIM, 256), 256, 0, stream>>>(wg, wgb, DIM, DIM);
    f32_to_bf16_T_kernel<<<cdiv(DIM * 2 * DIM, 256), 256, 0, stream>>>(wa, wab, DIM, 2 * DIM);
    f32_to_bf16_T_kernel<<<cdiv(DIM * DIM, 256), 256, 0, stream>>>(wo, wob, DIM, DIM);

    // 2) projection GEMMs (v_wmma_f32_16x16x32_bf16)
    const int tiles256 = (SEQ / 16) * (DIM / 16);       // 1024 tiles
    const int tiles512 = (SEQ / 16) * (2 * DIM / 16);   // 2048 tiles
    gemm_bf16_wmma_kernel<<<cdiv(tiles256, 8), 256, 0, stream>>>(xb, wqb, nullptr, qf, SEQ, DIM, DIM);
    gemm_bf16_wmma_kernel<<<cdiv(tiles256, 8), 256, 0, stream>>>(xb, wkb, nullptr, kf, SEQ, DIM, DIM);
    gemm_bf16_wmma_kernel<<<cdiv(tiles256, 8), 256, 0, stream>>>(xb, wvb, nullptr, vf, SEQ, DIM, DIM);
    gemm_bf16_wmma_kernel<<<cdiv(tiles256, 8), 256, 0, stream>>>(xb, wgb, nullptr, gf, SEQ, DIM, DIM);
    gemm_bf16_wmma_kernel<<<cdiv(tiles512, 8), 256, 0, stream>>>(xb, wab, nullptr, af, SEQ, 2 * DIM, DIM);

    // 3) complex gate decomposition
    gate_decompose_kernel<<<cdiv(SEQ * DIM, 256), 256, 0, stream>>>(af, amr, ami);

    // 4) gated linear recurrence: 16 d-blocks in parallel, states in VGPRs
    gated_scan_kernel<<<16, 256, 0, stream>>>(kf, vf, qf, amr, ami, part);

    // 5) reduce partials + GroupNorm + silu gate -> bf16 z
    reduce_gn_gate_kernel<<<cdiv(SEQ * GROUPS, 256), 256, 0, stream>>>(part, gf, gnw, gnb, zb);

    // 6) output GEMM with bias (WMMA)
    gemm_bf16_wmma_kernel<<<cdiv(tiles256, 8), 256, 0, stream>>>(zb, wob, bias, out, SEQ, DIM, DIM);
}
